// MemoryAsContextTransformer_14379550507471
// MI455X (gfx1250) — matrix operand order repaired and checked by reading, not compiled, not run on hardware
//
#include <hip/hip_runtime.h>
#include <hip/hip_bf16.h>

// ---------------------------------------------------------------------------
// MemoryAsContext transformer forward for gfx1250 (MI455X).
// bf16 WMMA everywhere; fragment-ready weight swizzle (B streamed from L2);
// TDM (tensor_load_to_lds) double-buffered A-tile staging.
// ---------------------------------------------------------------------------

typedef __attribute__((ext_vector_type(16))) __bf16 bf16x16;
typedef __attribute__((ext_vector_type(8)))  float  f32x8;
typedef __attribute__((ext_vector_type(4)))  unsigned u32x4;
typedef __attribute__((ext_vector_type(8)))  int      i32x8;
typedef __attribute__((ext_vector_type(4)))  int      i32x4;

#define NB  2
#define NS  4096
#define NBS 8192            // B*S tokens
#define NDIM 512
#define NHEADS 8
#define NDH 64
#define NSEG 512
#define NW  8
#define NPM 4
#define KVCH 17             // key chunks of 32 (PM+SEG padded 516 -> 544)
#define NVOCAB 32000
#define FFI 1365
#define FFI_PAD 1408        // 11*128
#define FF2 2730
#define FF2_PAD 2816        // 22*128

#if __has_builtin(__builtin_amdgcn_tensor_load_to_lds) && \
    __has_builtin(__builtin_amdgcn_s_wait_tensorcnt)
#define HAVE_TDM 1
#else
#define HAVE_TDM 0
#endif

__device__ __forceinline__ __bf16 f2bf(float f) {
    unsigned u = __builtin_bit_cast(unsigned, f);
    unsigned r = (u + 0x7FFFu + ((u >> 16) & 1u)) >> 16;
    return __builtin_bit_cast(__bf16, (unsigned short)r);
}
__device__ __forceinline__ float bf2f(__bf16 h) {
    unsigned short s = __builtin_bit_cast(unsigned short, h);
    unsigned u = ((unsigned)s) << 16;
    return __builtin_bit_cast(float, u);
}
__device__ __forceinline__ f32x8 zero8() {
    f32x8 z;
#pragma unroll
    for (int i = 0; i < 8; ++i) z[i] = 0.0f;
    return z;
}
__device__ __forceinline__ f32x8 wmma_bf16(bf16x16 a, bf16x16 b, f32x8 c) {
    return __builtin_amdgcn_wmma_f32_16x16x32_bf16(false, a, false, b,
                                                   (short)0, c, false, false);
}

// Load a 16-element bf16 fragment from two aligned 16B chunks (LDS or global).
struct B32x2 { uint4 lo; uint4 hi; };
__device__ __forceinline__ bf16x16 ld_frag(const __bf16* p0, const __bf16* p1) {
    B32x2 q;
    q.lo = *(const uint4*)p0;
    q.hi = *(const uint4*)p1;
    return __builtin_bit_cast(bf16x16, q);
}

#if HAVE_TDM
// Issue a 2-D TDM load: tileH rows x tileW bf16 elements, row stride
// strideElems, into LDS at ldsByteAddr. Descriptor per CDNA5 ISA §8.3/8.4.
__device__ __forceinline__ void tdm_load_tile_2d(unsigned ldsByteAddr,
                                                 const void* gaddr,
                                                 unsigned tileW, unsigned tileH,
                                                 unsigned tensorW, unsigned tensorH,
                                                 unsigned long long strideElems) {
    unsigned long long ga = (unsigned long long)gaddr;
    u32x4 g0;
    g0[0] = 1u;                                   // count=1, user descriptor
    g0[1] = ldsByteAddr;                          // lds_addr (bytes)
    g0[2] = (unsigned)ga;                         // global_addr[31:0]
    g0[3] = (unsigned)((ga >> 32) & 0x01FFFFFFull) | 0x80000000u; // [56:32]|type=2
    i32x8 g1;
    g1[0] = (int)(1u << 16);                      // data_size=1 -> 2 bytes
    g1[1] = (int)((tensorW & 0xFFFFu) << 16);     // tensor_dim0[15:0]
    g1[2] = (int)(((tensorW >> 16) & 0xFFFFu) | ((tensorH & 0xFFFFu) << 16));
    g1[3] = (int)(((tensorH >> 16) & 0xFFFFu) | ((tileW & 0xFFFFu) << 16));
    g1[4] = (int)(tileH & 0xFFFFu);               // tile_dim1 (tile_dim2 = 0)
    g1[5] = (int)(unsigned)(strideElems & 0xFFFFFFFFull);   // dim0_stride lo
    g1[6] = (int)(unsigned)((strideElems >> 32) & 0xFFFFull);
    g1[7] = 0;
    i32x4 gz; gz[0] = gz[1] = gz[2] = gz[3] = 0;
#if defined(__clang_major__) && (__clang_major__ >= 23)
    i32x8 gz8;
#pragma unroll
    for (int i = 0; i < 8; ++i) gz8[i] = 0;
    __builtin_amdgcn_tensor_load_to_lds(g0, g1, gz, gz, gz8, 0);
#else
    __builtin_amdgcn_tensor_load_to_lds(g0, g1, gz, gz, 0);
#endif
}
#endif

// ---------------------------------------------------------------------------
// Fragment-layout index helpers for attention K / V buffers.
// K used as B-matrix of QK^T:  [bhw][chunk][hh(2)][ks(2)][lane(32)][16]
// V used as B-matrix of P*V :  [bhw][chunk][nt(4)][lane(32)][16]
// ---------------------------------------------------------------------------
__device__ __forceinline__ size_t kf_index(int bhw, int j, int dh) {
    int c = j >> 5, jin = j & 31;
    int hh = jin >> 4, nl = jin & 15;
    int ks = dh >> 5, din = dh & 31;
    int lane = nl | ((din >> 4) << 4);
    return (((((size_t)bhw * KVCH + c) * 2 + hh) * 2 + ks) * 32 + lane) * 16 + (din & 15);
}
__device__ __forceinline__ size_t vf_index(int bhw, int j, int dh) {
    int c = j >> 5, jin = j & 31;
    int lane = (dh & 15) | ((jin >> 4) << 4);
    return ((((size_t)bhw * KVCH + c) * 4 + (dh >> 4)) * 32 + lane) * 16 + (jin & 15);
}

// ---------------------------------------------------------------------------
// fp32 [K,N] weights -> bf16 WMMA-B-fragment layout with zero padding:
// dst[((kchunk*(Np/16) + n/16)*32 + lane)*16 + e]
// ---------------------------------------------------------------------------
__global__ void k_cvt_fragB(const float* __restrict__ src, __bf16* __restrict__ dst,
                            int srcK, int srcN, int Kp, int Np) {
    long total = (long)Kp * Np;
    for (long i = blockIdx.x * (long)blockDim.x + threadIdx.x; i < total;
         i += (long)gridDim.x * blockDim.x) {
        int k = (int)(i / Np), n = (int)(i % Np);
        float v = (k < srcK && n < srcN) ? src[(long)k * srcN + n] : 0.0f;
        int kchunk = k >> 5, kin = k & 31;
        int lane = (n & 15) | ((kin >> 4) << 4);
        size_t dj = (((size_t)kchunk * (Np >> 4) + (n >> 4)) * 32 + lane) * 16 + (kin & 15);
        dst[dj] = f2bf(v);
    }
}

// ---------------------------------------------------------------------------
// Embedding
// ---------------------------------------------------------------------------
__global__ __launch_bounds__(256)
void k_embed(const int* __restrict__ toks, const float* __restrict__ TE,
             const float* __restrict__ PE, float* __restrict__ X) {
    long t = blockIdx.x;
    int tok = toks[t];
    int s = (int)(t & (NS - 1));
    for (int c = threadIdx.x; c < NDIM; c += 256)
        X[t * NDIM + c] = TE[(size_t)tok * NDIM + c] + PE[(size_t)s * NDIM + c];
}

// ---------------------------------------------------------------------------
// RMSNorm (dim 512) -> bf16
// ---------------------------------------------------------------------------
__global__ __launch_bounds__(256)
void k_rmsnorm(const float* __restrict__ X, const float* __restrict__ Wn,
               __bf16* __restrict__ XN) {
    __shared__ float red[8];
    long r = blockIdx.x;
    int tid = threadIdx.x;
    float v0 = X[r * NDIM + tid];
    float v1 = X[r * NDIM + tid + 256];
    float ss = v0 * v0 + v1 * v1;
#pragma unroll
    for (int off = 16; off > 0; off >>= 1) ss += __shfl_xor(ss, off, 32);
    if ((tid & 31) == 0) red[tid >> 5] = ss;
    __syncthreads();
    float tot = 0.0f;
#pragma unroll
    for (int i = 0; i < 8; ++i) tot += red[i];
    float inv = rsqrtf(tot * (1.0f / NDIM) + 1e-6f);
    XN[r * NDIM + tid]       = f2bf(v0 * inv * Wn[tid]);
    XN[r * NDIM + tid + 256] = f2bf(v1 * inv * Wn[tid + 256]);
}

// ---------------------------------------------------------------------------
// bf16 GEMM:  C[M,N] = A[M,K] @ B[K,N] (+bias) (+residual), f32 out.
// A: row-major bf16, staged to LDS via TDM (double-buffered).
// B: fragment-ready bf16 (streamed from L2, no LDS).
// Tile 128x128, 8 waves of 32x64; M,N multiples of 128; K multiple of 32.
// ---------------------------------------------------------------------------
__global__ __launch_bounds__(256)
void k_gemm_bf16(const __bf16* __restrict__ A, int lda,
                 const __bf16* __restrict__ Bf,
                 float* __restrict__ C, int ldc,
                 const float* __restrict__ bias,
                 const float* __restrict__ residual,
                 int M, int N, int K) {
    __shared__ __bf16 Alds[2][128 * 32];

    const int tid  = threadIdx.x;
    const int lane = tid & 31, wid = tid >> 5;
    const int mo = (wid & 3) * 32, no = (wid >> 2) * 64;
    const int m0 = blockIdx.y * 128, n0 = blockIdx.x * 128;
    const int ml = lane & 15, khalf = lane >> 4;
    const int nT = N >> 4;
    const int nk = K >> 5;

    f32x8 acc[2][4];
#pragma unroll
    for (int mt = 0; mt < 2; ++mt)
#pragma unroll
        for (int nt = 0; nt < 4; ++nt) acc[mt][nt] = zero8();

    auto compute = [&](const __bf16* Abuf, int kchunk) {
        bf16x16 af[2], bfv[4];
#pragma unroll
        for (int mt = 0; mt < 2; ++mt) {
            const __bf16* row = Abuf + (mo + mt * 16 + ml) * 32;
            af[mt] = ld_frag(row + khalf * 16, row + khalf * 16 + 8);
        }
        const __bf16* bbase =
            Bf + ((size_t)kchunk * nT + ((n0 + no) >> 4)) * (32 * 16) + lane * 16;
#pragma unroll
        for (int nt = 0; nt < 4; ++nt) {
            const __bf16* p = bbase + nt * (32 * 16);
            bfv[nt] = ld_frag(p, p + 8);
        }
        if (kchunk + 1 < nk)   // gfx1250 global_prefetch_b8 of next B fragments
            __builtin_prefetch(bbase + (size_t)nT * (32 * 16), 0, 1);
#pragma unroll
        for (int mt = 0; mt < 2; ++mt)
#pragma unroll
            for (int nt = 0; nt < 4; ++nt)
                acc[mt][nt] = wmma_bf16(af[mt], bfv[nt], acc[mt][nt]);
    };

#if HAVE_TDM
    // Double-buffered TDM pipeline: wave 0 drives the Tensor Data Mover.
    if (wid == 0)
        tdm_load_tile_2d((unsigned)(unsigned long long)(uintptr_t)&Alds[0][0],
                         A + (size_t)m0 * lda, 32, 128,
                         (unsigned)K, (unsigned)(M - m0),
                         (unsigned long long)lda);
    for (int kt = 0; kt < nk; ++kt) {
        int cur = kt & 1;
        if (wid == 0) __builtin_amdgcn_s_wait_tensorcnt(0);
        __syncthreads();    // tile kt visible; everyone done with other buffer
        if (wid == 0 && kt + 1 < nk)
            tdm_load_tile_2d((unsigned)(unsigned long long)(uintptr_t)&Alds[cur ^ 1][0],
                             A + (size_t)m0 * lda + (kt + 1) * 32, 32, 128,
                             (unsigned)(K - (kt + 1) * 32), (unsigned)(M - m0),
                             (unsigned long long)lda);
        compute(&Alds[cur][0], kt);
    }
#else
    const int arow = tid >> 1, ahalf = tid & 1;  // 128 rows x 2 halves of 16
    for (int kt = 0; kt < nk; ++kt) {
        __syncthreads();
        const uint4* s = (const uint4*)(A + (size_t)(m0 + arow) * lda + kt * 32 + ahalf * 16);
        uint4* d = (uint4*)(&Alds[0][0] + arow * 32 + ahalf * 16);
        d[0] = s[0]; d[1] = s[1];
        __syncthreads();
        compute(&Alds[0][0], kt);
    }
#endif

    const int hg = lane >> 4;
#pragma unroll
    for (int mt = 0; mt < 2; ++mt) {
        int mbase = m0 + mo + mt * 16 + hg * 8;
#pragma unroll
        for (int nt = 0; nt < 4; ++nt) {
            int n = n0 + no + nt * 16 + ml;
#pragma unroll
            for (int v = 0; v < 8; ++v) {
                size_t idx = (size_t)(mbase + v) * ldc + n;
                float val = acc[mt][nt][v];
                if (bias)     val += bias[n];
                if (residual) val += residual[idx];
                C[idx] = val;
            }
        }
    }
}

// ---------------------------------------------------------------------------
// vmix = sigmoid(xn @ vmix_w + b) : one wave per (token, head)
// ---------------------------------------------------------------------------
__global__ __launch_bounds__(256)
void k_vmix(const __bf16* __restrict__ XN, const float* __restrict__ Wm,
            const float* __restrict__ bm, float* __restrict__ MIX) {
    long gw = (blockIdx.x * (long)blockDim.x + threadIdx.x) >> 5;
    int lane = threadIdx.x & 31;
    if (gw >= (long)NBS * NHEADS) return;
    int h = (int)(gw & (NHEADS - 1));
    long t = gw >> 3;
    float acc = 0.0f;
    for (int c = lane; c < NDIM; c += 32)
        acc += bf2f(XN[t * NDIM + c]) * Wm[c * NHEADS + h];
#pragma unroll
    for (int off = 16; off > 0; off >>= 1) acc += __shfl_xor(acc, off, 32);
    if (lane == 0)
        MIX[t * NHEADS + h] = 1.0f / (1.0f + __expf(-(acc + bm[h])));
}

// ---------------------------------------------------------------------------
// QKV post: RoPE(q,k), value-residual mix, scatter K/V into fragment layout.
// ---------------------------------------------------------------------------
__global__ __launch_bounds__(256)
void k_qkv_post(const float* __restrict__ QKV, const float* __restrict__ MIX,
                float* __restrict__ VRES, __bf16* __restrict__ QR,
                __bf16* __restrict__ KF, __bf16* __restrict__ VF, int layer) {
    const long total = (long)NBS * NHEADS * (NDH / 2);
    for (long idx = blockIdx.x * (long)blockDim.x + threadIdx.x; idx < total;
         idx += (long)gridDim.x * blockDim.x) {
        int i = (int)(idx & 31);              // rotary pair index
        int h = (int)((idx >> 5) & 7);
        long t = idx >> 8;                    // token
        int b = (int)(t >> 12);
        int s = (int)(t & (NS - 1));
        int w = s >> 9, sq = s & (NSEG - 1);
        int bhw = (b * NHEADS + h) * NW + w;

        const float* base = QKV + t * (3 * NDIM) + h * NDH + 2 * i;
        float q1 = base[0],    q2 = base[1];
        float k1 = base[512],  k2 = base[513];
        float v1 = base[1024], v2 = base[1025];

        float freq = powf(10000.0f, -(float)(2 * i) / (float)NDH);
        float ang = (float)s * freq;
        float cs = __cosf(ang), sn = __sinf(ang);

        size_t qoff = ((size_t)(b * NHEADS + h) * NS + s) * NDH + 2 * i;
        QR[qoff]     = f2bf(q1 * cs - q2 * sn);
        QR[qoff + 1] = f2bf(q1 * sn + q2 * cs);

        int j = NPM + sq;                     // key row
        size_t kidx = kf_index(bhw, j, 2 * i);  // pair is contiguous (e even)
        KF[kidx]     = f2bf(k1 * cs - k2 * sn);
        KF[kidx + 1] = f2bf(k1 * sn + k2 * cs);

        if (layer == 0) {
            VRES[qoff] = v1; VRES[qoff + 1] = v2;
        } else {
            float m = MIX[t * NHEADS + h];
            v1 = v1 + (VRES[qoff]     - v1) * m;
            v2 = v2 + (VRES[qoff + 1] - v2) * m;
        }
        VF[vf_index(bhw, j, 2 * i)]     = f2bf(v1);
        VF[vf_index(bhw, j, 2 * i + 1)] = f2bf(v2);
    }
}

// ---------------------------------------------------------------------------
// Persistent memory -> K/V fragment rows 0..3; zero pad rows 516..543.
// ---------------------------------------------------------------------------
__global__ __launch_bounds__(256)
void k_pack_pm(const float* __restrict__ PMw, __bf16* __restrict__ KF,
               __bf16* __restrict__ VF) {
    const long total = (long)NB * NHEADS * NW * 32 * NDH;
    for (long idx = blockIdx.x * (long)blockDim.x + threadIdx.x; idx < total;
         idx += (long)gridDim.x * blockDim.x) {
        int e   = (int)(idx & 63);
        int rr  = (int)((idx >> 6) & 31);
        int bhw = (int)(idx >> 11);
        int h = (bhw >> 3) & 7;
        int j = (rr < NPM) ? rr : (512 + rr);     // 0..3 or 516..543
        float kv = 0.0f, vv = 0.0f;
        if (rr < NPM) {
            kv = PMw[((0 * NHEADS + h) * NPM + rr) * NDH + e];
            vv = PMw[((1 * NHEADS + h) * NPM + rr) * NDH + e];
        }
        KF[kf_index(bhw, j, e)] = f2bf(kv);
        VF[vf_index(bhw, j, e)] = f2bf(vv);
    }
}

// ---------------------------------------------------------------------------
// Segmented flash attention, fully barrier-free (per-wave independent).
// Grid: B*H*W*8 blocks of 128 threads; each wave = 16 query rows.
// K/V come pre-swizzled in fragment layout -> 2x b128 loads per fragment.
// ---------------------------------------------------------------------------
__global__ __launch_bounds__(128)
void k_attention(const __bf16* __restrict__ QR, const __bf16* __restrict__ KF,
                 const __bf16* __restrict__ VF, __bf16* __restrict__ Ob) {
    __shared__ __bf16 Pl[4][16 * 32];

    const int blk = blockIdx.x;
    const int qt = blk & 7;
    const int w  = (blk >> 3) & 7;
    const int h  = (blk >> 6) & 7;
    const int b  = blk >> 9;
    const int tid = threadIdx.x, lane = tid & 31, wid = tid >> 5;
    const int qrow0 = qt * 64 + wid * 16;
    const int ml = lane & 15, hg = lane >> 4;
    const int bhw = (b * NHEADS + h) * NW + w;

    // Q fragments (A-layout): two aligned 16B chunks per k-step
    bf16x16 aq[2];
    {
        const __bf16* qrow =
            QR + ((size_t)(b * NHEADS + h) * NS + w * NSEG + qrow0 + ml) * NDH;
#pragma unroll
        for (int ks = 0; ks < 2; ++ks)
            aq[ks] = ld_frag(qrow + ks * 32 + hg * 16, qrow + ks * 32 + hg * 16 + 8);
    }

    f32x8 o[4];
#pragma unroll
    for (int nt = 0; nt < 4; ++nt) o[nt] = zero8();
    float mrun[8], lrun[8];
#pragma unroll
    for (int v = 0; v < 8; ++v) { mrun[v] = -1e30f; lrun[v] = 0.0f; }

    const int nchunks = (qrow0 + 51) >> 5;   // covers keys j <= NPM+qrow0+15

    for (int c = 0; c < nchunks; ++c) {
        // scores S[16x32] = two 16x16 WMMA tiles, K fragments from global
        f32x8 sfrag[2];
#pragma unroll
        for (int hh = 0; hh < 2; ++hh) {
            f32x8 s = zero8();
#pragma unroll
            for (int ks = 0; ks < 2; ++ks) {
                const __bf16* p = KF +
                    (((((size_t)bhw * KVCH + c) * 2 + hh) * 2 + ks) * 32 + lane) * 16;
                s = wmma_bf16(aq[ks], ld_frag(p, p + 8), s);
            }
            sfrag[hh] = s;
        }

        // online softmax (rows per C-layout: row = v + 8*hg)
#pragma unroll
        for (int v = 0; v < 8; ++v) {
            int qi = qrow0 + hg * 8 + v;
            float sc0 = sfrag[0][v] * 0.125f;
            float sc1 = sfrag[1][v] * 0.125f;
            int j0 = c * 32 + ml, j1 = c * 32 + 16 + ml;
            if (!(j0 < NPM || (j0 - NPM) <= qi)) sc0 = -1e30f;
            if (!(j1 < NPM || (j1 - NPM) <= qi)) sc1 = -1e30f;
            float rmax = fmaxf(sc0, sc1);
#pragma unroll
            for (int off = 8; off > 0; off >>= 1)
                rmax = fmaxf(rmax, __shfl_xor(rmax, off, 16));
            float mnew = fmaxf(mrun[v], rmax);
            float corr = __expf(mrun[v] - mnew);
            float p0 = __expf(sc0 - mnew);
            float p1 = __expf(sc1 - mnew);
            float psum = p0 + p1;
#pragma unroll
            for (int off = 8; off > 0; off >>= 1) psum += __shfl_xor(psum, off, 16);
            lrun[v] = lrun[v] * corr + psum;
            mrun[v] = mnew;
#pragma unroll
            for (int nt = 0; nt < 4; ++nt) o[nt][v] *= corr;
            int prow = hg * 8 + v;
            Pl[wid][prow * 32 + ml]      = f2bf(p0);
            Pl[wid][prow * 32 + 16 + ml] = f2bf(p1);
        }

        // O += P.V  (P A-fragment via per-wave LDS; V fragments from global)
        const __bf16* prow = &Pl[wid][0] + ml * 32;
        bf16x16 ap = ld_frag(prow + hg * 16, prow + hg * 16 + 8);
#pragma unroll
        for (int nt = 0; nt < 4; ++nt) {
            const __bf16* p = VF +
                ((((size_t)bhw * KVCH + c) * 4 + nt) * 32 + lane) * 16;
            o[nt] = wmma_bf16(ap, ld_frag(p, p + 8), o[nt]);
        }
    }

    // normalize and write [B,S, h*64+dh] bf16 for the Wout GEMM
#pragma unroll
    for (int v = 0; v < 8; ++v) {
        float inv = 1.0f / lrun[v];
        int srow = w * NSEG + qrow0 + hg * 8 + v;
        size_t base = ((size_t)b * NS + srow) * (NHEADS * NDH) + h * NDH;
#pragma unroll
        for (int nt = 0; nt < 4; ++nt)
            Ob[base + nt * 16 + ml] = f2bf(o[nt][v] * inv);
    }
}

// ---------------------------------------------------------------------------
// GEGLU: FFA = silu(g) * a, bf16 out, padded cols zeroed.
// ---------------------------------------------------------------------------
__global__ void k_geglu(const float* __restrict__ H1, const float* __restrict__ b1,
                        __bf16* __restrict__ FFA) {
    const long total = (long)NBS * FFI_PAD;
    for (long i = blockIdx.x * (long)blockDim.x + threadIdx.x; i < total;
         i += (long)gridDim.x * blockDim.x) {
        long t = i / FFI_PAD;
        int j = (int)(i % FFI_PAD);
        float out = 0.0f;
        if (j < FFI) {
            float a = H1[t * FF2_PAD + j] + b1[j];
            float g = H1[t * FF2_PAD + FFI + j] + b1[FFI + j];
            out = a * (g / (1.0f + __expf(-g)));
        }
        FFA[i] = f2bf(out);
    }
}

// ---------------------------------------------------------------------------
// Host launcher
// ---------------------------------------------------------------------------
extern "C" void kernel_launch(void* const* d_in, const int* in_sizes, int n_in,
                              void* d_out, int out_size, void* d_ws, size_t ws_size,
                              hipStream_t stream) {
    (void)in_sizes; (void)n_in; (void)out_size; (void)ws_size;

    const int*   tokens      = (const int*)  d_in[0];
    const float* tok_emb     = (const float*)d_in[1];
    const float* pos_emb     = (const float*)d_in[2];
    const float* attn_norm_w = (const float*)d_in[3];
    const float* Wqkv        = (const float*)d_in[4];
    const float* persist_mem = (const float*)d_in[5];
    const float* Wout        = (const float*)d_in[6];
    const float* vmix_w      = (const float*)d_in[7];
    const float* vmix_b      = (const float*)d_in[8];
    const float* ff_norm_w   = (const float*)d_in[9];
    const float* ff_w1       = (const float*)d_in[10];
    const float* ff_b1       = (const float*)d_in[11];
    const float* ff_w2       = (const float*)d_in[12];
    const float* ff_b2       = (const float*)d_in[13];
    const float* final_norm_w= (const float*)d_in[14];
    const float* w_logits    = (const float*)d_in[15];
    float* out = (float*)d_out;

    char* wsb = (char*)d_ws;
    size_t off = 0;
    auto take = [&](size_t bytes) -> void* {
        void* p = wsb + off;
        off += (bytes + 255) & ~(size_t)255;
        return p;
    };

    float*  X     = (float*) take((size_t)NBS * NDIM * 4);
    __bf16* XN    = (__bf16*)take((size_t)NBS * NDIM * 2);
    float*  QKV   = (float*) take((size_t)NBS * 3 * NDIM * 4);
    __bf16* QR    = (__bf16*)take((size_t)NB * NHEADS * NS * NDH * 2);
    __bf16* KF    = (__bf16*)take((size_t)NB * NHEADS * NW * KVCH * 32 * NDH * 2);
    __bf16* VF    = (__bf16*)take((size_t)NB * NHEADS * NW * KVCH * 32 * NDH * 2);
    float*  VRES  = (float*) take((size_t)NB * NHEADS * NS * NDH * 4);
    float*  MIX   = (float*) take((size_t)NBS * NHEADS * 4);
    __bf16* ATTNB = (__bf16*)take((size_t)NBS * NDIM * 2);
    float*  H1    = (float*) take((size_t)NBS * FF2_PAD * 4);
    __bf16* FFA   = (__bf16*)take((size_t)NBS * FFI_PAD * 2);
    __bf16* WQKVb = (__bf16*)take((size_t)2 * NDIM * 3 * NDIM * 2);
    __bf16* WOUTb = (__bf16*)take((size_t)2 * NDIM * NDIM * 2);
    __bf16* WF1b  = (__bf16*)take((size_t)2 * NDIM * FF2_PAD * 2);
    __bf16* WF2b  = (__bf16*)take((size_t)2 * FFI_PAD * NDIM * 2);
    __bf16* WLOGb = (__bf16*)take((size_t)NDIM * NVOCAB * 2);

    // ---- weight conversion: fp32 -> bf16 fragment layout (per matrix) ----
    for (int d = 0; d < 2; ++d) {
        k_cvt_fragB<<<2048, 256, 0, stream>>>(Wqkv + (size_t)d * NDIM * 3 * NDIM,
                                              WQKVb + (size_t)d * NDIM * 3 * NDIM,
                                              NDIM, 3 * NDIM, NDIM, 3 * NDIM);
        k_cvt_fragB<<<2048, 256, 0, stream>>>(Wout + (size_t)d * NDIM * NDIM,
                                              WOUTb + (size_t)d * NDIM * NDIM,
                                              NDIM, NDIM, NDIM, NDIM);
        k_cvt_fragB<<<2048, 256, 0, stream>>>(ff_w1 + (size_t)d * NDIM * FF2,
                                              WF1b + (size_t)d * NDIM * FF2_PAD,
                                              NDIM, FF2, NDIM, FF2_PAD);
        k_cvt_fragB<<<2048, 256, 0, stream>>>(ff_w2 + (size_t)d * FFI * NDIM,
                                              WF2b + (size_t)d * FFI_PAD * NDIM,
                                              FFI, NDIM, FFI_PAD, NDIM);
    }
    k_cvt_fragB<<<4096, 256, 0, stream>>>(w_logits, WLOGb, NDIM, NVOCAB, NDIM, NVOCAB);

    // ---- embedding ----
    k_embed<<<NBS, 256, 0, stream>>>(tokens, tok_emb, pos_emb, X);

    for (int d = 0; d < 2; ++d) {
        // attention block
        k_rmsnorm<<<NBS, 256, 0, stream>>>(X, attn_norm_w + (size_t)d * NDIM, XN);
        {
            dim3 g(3 * NDIM / 128, NBS / 128);
            k_gemm_bf16<<<g, 256, 0, stream>>>(XN, NDIM,
                                               WQKVb + (size_t)d * NDIM * 3 * NDIM,
                                               QKV, 3 * NDIM, nullptr, nullptr,
                                               NBS, 3 * NDIM, NDIM);
        }
        if (d > 0)
            k_vmix<<<(NBS * NHEADS * 32 + 255) / 256, 256, 0, stream>>>(
                XN, vmix_w + (size_t)d * NDIM * NHEADS, vmix_b + (size_t)d * NHEADS, MIX);

        k_qkv_post<<<4096, 256, 0, stream>>>(QKV, MIX, VRES, QR, KF, VF, d);
        k_pack_pm<<<1024, 256, 0, stream>>>(
            persist_mem + (size_t)d * 2 * NHEADS * NPM * NDH, KF, VF);

        k_attention<<<NB * NHEADS * NW * 8, 128, 0, stream>>>(QR, KF, VF, ATTNB);

        {
            dim3 g(NDIM / 128, NBS / 128);
            k_gemm_bf16<<<g, 256, 0, stream>>>(ATTNB, NDIM,
                                               WOUTb + (size_t)d * NDIM * NDIM,
                                               X, NDIM, nullptr, X,
                                               NBS, NDIM, NDIM);
        }

        // GEGLU feedforward
        k_rmsnorm<<<NBS, 256, 0, stream>>>(X, ff_norm_w + (size_t)d * NDIM, XN);
        {
            dim3 g(FF2_PAD / 128, NBS / 128);
            k_gemm_bf16<<<g, 256, 0, stream>>>(XN, NDIM,
                                               WF1b + (size_t)d * NDIM * FF2_PAD,
                                               H1, FF2_PAD, nullptr, nullptr,
                                               NBS, FF2_PAD, NDIM);
        }
        k_geglu<<<8192, 256, 0, stream>>>(H1, ff_b1 + (size_t)d * FF2, FFA);
        {
            dim3 g(NDIM / 128, NBS / 128);
            k_gemm_bf16<<<g, 256, 0, stream>>>(FFA, FFI_PAD,
                                               WF2b + (size_t)d * FFI_PAD * NDIM,
                                               X, NDIM, ff_b2 + (size_t)d * NDIM, X,
                                               NBS, NDIM, FFI_PAD);
        }
    }

    // ---- final norm + logits ----
    k_rmsnorm<<<NBS, 256, 0, stream>>>(X, final_norm_w, XN);
    {
        dim3 g(NVOCAB / 128, NBS / 128);
        k_gemm_bf16<<<g, 256, 0, stream>>>(XN, NDIM, WLOGb,
                                           out, NVOCAB, nullptr, nullptr,
                                           NBS, NVOCAB, NDIM);
    }
}
